// GeoAB_62508954026302
// MI455X (gfx1250) — matrix operand order repaired and checked by reading, not compiled, not run on hardware
//
#include <hip/hip_runtime.h>
#include <math.h>

// ---------------------------------------------------------------------------
// Types for WMMA (gfx1250): v16bf A/B fragments, v8f C/D accumulator
// ---------------------------------------------------------------------------
typedef __bf16 v16bf __attribute__((ext_vector_type(16)));
typedef __bf16 v8bf  __attribute__((ext_vector_type(8)));
typedef float  v8f   __attribute__((ext_vector_type(8)));

__device__ __forceinline__ v16bf cat16(v8bf lo, v8bf hi) {
  return __builtin_shufflevector(lo, hi, 0,1,2,3,4,5,6,7,8,9,10,11,12,13,14,15);
}

__device__ __forceinline__ unsigned short f2bf(float f) {
  unsigned int u = __float_as_uint(f);
  u += 0x7FFFu + ((u >> 16) & 1u);     // round-to-nearest-even
  return (unsigned short)(u >> 16);
}

// ---------------------------------------------------------------------------
// Pack NCHW f32 -> zero-padded NHWC bf16 (B, 162, 162, C)
// ---------------------------------------------------------------------------
__global__ void pack_pad_nhwc_bf16(const float* __restrict__ src,
                                   unsigned short* __restrict__ dst, int C) {
  long idx = (long)blockIdx.x * blockDim.x + threadIdx.x;
  long total = (long)2 * 162 * 162 * C;
  if (idx >= total) return;
  int c = (int)(idx % C);
  long r = idx / C;
  int xx = (int)(r % 162); r /= 162;
  int yy = (int)(r % 162);
  int b  = (int)(r / 162);
  float v = 0.f;
  if (yy >= 1 && yy <= 160 && xx >= 1 && xx <= 160)
    v = src[(((size_t)b * C + c) * 160 + (yy - 1)) * 160 + (xx - 1)];
  dst[idx] = f2bf(v);
}

// Pack weights (Cout, 192, 3, 3) f32 -> (Cout, 1728) bf16, K = kpos*192 + ic
__global__ void pack_w_bf16(const float* __restrict__ w,
                            unsigned short* __restrict__ dst, int Cout) {
  int idx = blockIdx.x * blockDim.x + threadIdx.x;
  if (idx >= Cout * 1728) return;
  int oc = idx / 1728;
  int k  = idx % 1728;
  int kpos = k / 192;
  int ic   = k - kpos * 192;
  int ky = kpos / 3, kx = kpos - ky * 3;
  dst[idx] = f2bf(w[(((size_t)oc * 192 + ic) * 3 + ky) * 3 + kx]);
}

// ---------------------------------------------------------------------------
// Implicit-GEMM 3x3 conv via v_wmma_f32_16x16x32_bf16.
// Block tile: M=64 (4 waves x 16 out-ch) x N=64 spatial, K = 192*9 (54 steps).
// B (im2col) tile is staged cooperatively in LDS each K-step (double-buffered)
// so all 4 waves share it -> 4x less L2 traffic than per-wave streaming.
// LDS row pitch 80B keeps the b128 ds_load fragment reads bank-conflict-free.
// ---------------------------------------------------------------------------
__global__ __launch_bounds__(128) void conv3x3_wmma_kernel(
    const unsigned short* __restrict__ xpad_u,   // (B,162,162,192) bf16
    const unsigned short* __restrict__ wpk_u,    // (Cout,1728) bf16
    const float* __restrict__ bias,              // (Cout)
    float* __restrict__ out,                     // (B,Cout,160,160)
    int Cout, int mblocks) {
  __shared__ __align__(16) unsigned short Bt[2][64 * 40];  // 2 x 64 rows x 80B
  const __bf16* wpk = reinterpret_cast<const __bf16*>(wpk_u);

  int mb = blockIdx.x % mblocks;
  int nt = blockIdx.x / mblocks;       // 0..799
  int n0 = nt * 64;
  int tid  = threadIdx.x;
  int wave = tid >> 5;
  int lane = tid & 31;
  int m0   = (mb * 4 + wave) * 16;
  int mrow  = lane & 15;
  int khalf = lane >> 4;

  // Cooperative-load assignment: thread -> (pixel row p, 16-elem half q)
  int p = tid >> 1;                    // 0..63 (LDS row / spatial pixel)
  int q = tid & 1;
  size_t pixBase;
  {
    int n = n0 + p;
    int b = n / 25600;
    int rr = n - b * 25600;
    int y = rr / 160;
    int x = rr - y * 160;
    pixBase = (((size_t)b * 162 + y) * 162 + x) * 192 + (size_t)q * 16;
  }

  // Output coordinates per N-subtile for the epilogue
  int bb[4], yy[4], xx[4];
#pragma unroll
  for (int t = 0; t < 4; ++t) {
    int n = n0 + t * 16 + (lane & 15);
    int b = n / 25600;
    int rr = n - b * 25600;
    int y = rr / 160;
    int x = rr - y * 160;
    bb[t] = b; yy[t] = y; xx[t] = x;
  }

  // A fragment per-lane K mapping: {khalf*8 + 0..7, 16 + khalf*8 + 0..7}
  const __bf16* aB = wpk + (size_t)(m0 + mrow) * 1728 + (size_t)(khalf * 8);

  v8f zero = {};
  v8f acc[4] = {zero, zero, zero, zero};

  int cur = 0;
  for (int ks = 0; ks < 54; ++ks) {
    int kpos = ks / 6;                 // kernel tap (0..8)
    int ic0  = (ks - kpos * 6) * 32;   // input-channel base within tap
    int ky = kpos / 3, kx = kpos - ky * 3;

    // Cooperative stage of 32K x 64N bf16 tile into LDS (32B per thread)
    const uint4* g = reinterpret_cast<const uint4*>(
        xpad_u + pixBase + ((size_t)ky * 162 + kx) * 192 + ic0);
    uint4 d0 = g[0];
    uint4 d1 = g[1];
    uint4* ls = reinterpret_cast<uint4*>(&Bt[cur][p * 40 + q * 16]);
    ls[0] = d0;
    ls[1] = d1;
    __syncthreads();

    const v8bf* ap = reinterpret_cast<const v8bf*>(aB + ks * 32);
    v16bf afrag = cat16(ap[0], ap[2]);
#pragma unroll
    for (int t = 0; t < 4; ++t) {
      // B fragment per-lane K mapping: {khalf*8 + 0..7, 16 + khalf*8 + 0..7}
      const unsigned short* bs = &Bt[cur][(t * 16 + (lane & 15)) * 40 + khalf * 8];
      const v8bf* bp = reinterpret_cast<const v8bf*>(bs);
      v16bf bfrag = cat16(bp[0], bp[2]);
      acc[t] = __builtin_amdgcn_wmma_f32_16x16x32_bf16(
          false, afrag, false, bfrag, (short)0, acc[t], false, false);
    }
    cur ^= 1;                          // double buffer: next store hits other half
  }

#pragma unroll
  for (int r = 0; r < 8; ++r) {
    int oc = m0 + r + khalf * 8;       // C/D layout: M = r + (lane>=16 ? 8 : 0)
    float bi = bias[oc];
#pragma unroll
    for (int t = 0; t < 4; ++t)
      out[(((size_t)bb[t] * Cout + oc) * 160 + yy[t]) * 160 + xx[t]] =
          acc[t][r] + bi;
  }
}

// ---------------------------------------------------------------------------
// CPB relative-position bias: (8, 25, 25) = 16*sigmoid(MLP table gather)
// ---------------------------------------------------------------------------
__global__ __launch_bounds__(128) void cpb_bias_kernel(
    const float* __restrict__ w1, const float* __restrict__ b1,
    const float* __restrict__ w2, float* __restrict__ bias) {
  __shared__ float tbl[81][8];
  int tid = threadIdx.x;
  if (tid < 81) {
    int a = tid / 9, b_ = tid % 9;
    float t0 = (float)(a - 4) * 2.0f;       // rel/4 * 8
    float t1 = (float)(b_ - 4) * 2.0f;
    const float inv_l28 = 1.0f / log2f(8.0f);
    float u0 = copysignf(log2f(fabsf(t0) + 1.0f) * inv_l28, t0);
    float u1 = copysignf(log2f(fabsf(t1) + 1.0f) * inv_l28, t1);
    float acc[8] = {0,0,0,0,0,0,0,0};
    for (int k = 0; k < 512; ++k) {
      float h = fmaxf(u0 * w1[2*k] + u1 * w1[2*k + 1] + b1[k], 0.f);
#pragma unroll
      for (int n = 0; n < 8; ++n) acc[n] += h * w2[n * 512 + k];
    }
#pragma unroll
    for (int n = 0; n < 8; ++n) tbl[tid][n] = acc[n];
  }
  __syncthreads();
  for (int e = tid; e < 8 * 625; e += 128) {
    int n = e / 625;
    int rest = e - n * 625;
    int i = rest / 25, j = rest - (rest / 25) * 25;
    int di = i / 5 - j / 5 + 4;
    int dj = i % 5 - j % 5 + 4;
    float v = tbl[di * 9 + dj][n];
    bias[e] = 16.0f / (1.0f + __expf(-v));
  }
}

// ---------------------------------------------------------------------------
// 5x5 window cosine attention. One block = one window; wave w = head.
// chanOff: k channel base in xp; dstOff: output channel base in yBuf.
// inShift/outShift implement the roll(-3)/roll(+2) of the shifted branch.
// ---------------------------------------------------------------------------
__global__ __launch_bounds__(256) void win_attn_kernel(
    const float* __restrict__ xp, const float* __restrict__ biasT,
    const float* __restrict__ lsc, float* __restrict__ yBuf,
    int chanOff, int dstOff, int inShift, int outShift) {
  __shared__ float kn[8][25][8];
  __shared__ float vv[8][25][8];
  int head = threadIdx.x >> 5;
  int lane = threadIdx.x & 31;
  int wid = blockIdx.x;
  int b = wid >> 10;
  int rem = wid & 1023;
  int wy = rem >> 5, wx = rem & 31;
  float scale = __expf(fminf(lsc[head], logf(100.f)));
  int i = lane;
  int sy = 0, sx = 0;
  if (i < 25) {
    int iy = i / 5, ix = i - iy * 5;
    sy = wy * 5 + iy; sx = wx * 5 + ix;
    int ysrc = (sy + inShift) % 160;
    int xsrc = (sx + inShift) % 160;
    float kr[8], vr[8], ss = 0.f;
#pragma unroll
    for (int c = 0; c < 8; ++c) {
      kr[c] = xp[(((size_t)b * 384 + chanOff + head * 8 + c) * 160 + ysrc) * 160 + xsrc];
      vr[c] = xp[(((size_t)b * 384 + chanOff + 64 + head * 8 + c) * 160 + ysrc) * 160 + xsrc];
      ss += kr[c] * kr[c];
    }
    float inv = 1.f / fmaxf(sqrtf(ss), 1e-12f);
#pragma unroll
    for (int c = 0; c < 8; ++c) { kn[head][i][c] = kr[c] * inv; vv[head][i][c] = vr[c]; }
  }
  __syncthreads();
  if (i < 25) {
    float qv[8];
#pragma unroll
    for (int c = 0; c < 8; ++c) qv[c] = kn[head][i][c];
    float m = -1e30f, l = 0.f;
    float acc[8] = {0,0,0,0,0,0,0,0};
    for (int j = 0; j < 25; ++j) {
      float s = 0.f;
#pragma unroll
      for (int c = 0; c < 8; ++c) s += qv[c] * kn[head][j][c];
      s = s * scale + biasT[(head * 25 + i) * 25 + j];
      float mn = fmaxf(m, s);
      float corr = __expf(m - mn);
      float pr = __expf(s - mn);
      l = l * corr + pr;
#pragma unroll
      for (int c = 0; c < 8; ++c) acc[c] = acc[c] * corr + pr * vv[head][j][c];
      m = mn;
    }
    int yd = (sy + outShift) % 160, xd = (sx + outShift) % 160;
    float invl = 1.f / l;
#pragma unroll
    for (int c = 0; c < 8; ++c)
      yBuf[(((size_t)b * 192 + dstOff + head * 8 + c) * 160 + yd) * 160 + xd] = acc[c] * invl;
  }
}

// ---------------------------------------------------------------------------
// Long-range attention, stage 1: along rows (x axis). Block=(b, y, head).
// ---------------------------------------------------------------------------
__global__ __launch_bounds__(160) void lr_attn_rows(
    const float* __restrict__ xp, const float* __restrict__ lrs,
    float* __restrict__ vtmp) {
  __shared__ float kn[160][8];
  __shared__ float vv[160][8];
  int bid = blockIdx.x;
  int head = bid % 8;
  int y = (bid / 8) % 160;
  int b = bid / (8 * 160);
  int t = threadIdx.x;                 // x
  float scale = __expf(fminf(lrs[head], logf(100.f)));
  float qv[8]; float ss = 0.f;
#pragma unroll
  for (int c = 0; c < 8; ++c) {
    float kv = xp[(((size_t)b * 384 + 256 + head * 8 + c) * 160 + y) * 160 + t];
    qv[c] = kv; ss += kv * kv;
  }
  float inv = 1.f / fmaxf(sqrtf(ss), 1e-12f);
#pragma unroll
  for (int c = 0; c < 8; ++c) {
    qv[c] *= inv; kn[t][c] = qv[c];
    vv[t][c] = xp[(((size_t)b * 384 + 320 + head * 8 + c) * 160 + y) * 160 + t];
  }
  __syncthreads();
  float m = -1e30f, l = 0.f;
  float acc[8] = {0,0,0,0,0,0,0,0};
  for (int j = 0; j < 160; ++j) {
    float s = 0.f;
#pragma unroll
    for (int c = 0; c < 8; ++c) s += qv[c] * kn[j][c];
    s *= scale;
    float mn = fmaxf(m, s);
    float corr = __expf(m - mn);
    float pr = __expf(s - mn);
    l = l * corr + pr;
#pragma unroll
    for (int c = 0; c < 8; ++c) acc[c] = acc[c] * corr + pr * vv[j][c];
    m = mn;
  }
  float invl = 1.f / l;
#pragma unroll
  for (int c = 0; c < 8; ++c)
    vtmp[((((size_t)b * 8 + head) * 160 + y) * 160 + t) * 8 + c] = acc[c] * invl;
}

// Stage 2: along columns (y axis). Block=(b, x, head). Writes y2 channels.
__global__ __launch_bounds__(160) void lr_attn_cols(
    const float* __restrict__ xp, const float* __restrict__ lrs,
    const float* __restrict__ vtmp, float* __restrict__ yBuf) {
  __shared__ float kn[160][8];
  __shared__ float vv[160][8];
  int bid = blockIdx.x;
  int head = bid % 8;
  int x = (bid / 8) % 160;
  int b = bid / (8 * 160);
  int t = threadIdx.x;                 // y
  float scale = __expf(fminf(lrs[head], logf(100.f)));
  float qv[8]; float ss = 0.f;
#pragma unroll
  for (int c = 0; c < 8; ++c) {
    float kv = xp[(((size_t)b * 384 + 256 + head * 8 + c) * 160 + t) * 160 + x];
    qv[c] = kv; ss += kv * kv;
  }
  float inv = 1.f / fmaxf(sqrtf(ss), 1e-12f);
#pragma unroll
  for (int c = 0; c < 8; ++c) {
    qv[c] *= inv; kn[t][c] = qv[c];
    vv[t][c] = vtmp[((((size_t)b * 8 + head) * 160 + t) * 160 + x) * 8 + c];
  }
  __syncthreads();
  float m = -1e30f, l = 0.f;
  float acc[8] = {0,0,0,0,0,0,0,0};
  for (int j = 0; j < 160; ++j) {
    float s = 0.f;
#pragma unroll
    for (int c = 0; c < 8; ++c) s += qv[c] * kn[j][c];
    s *= scale;
    float mn = fmaxf(m, s);
    float corr = __expf(m - mn);
    float pr = __expf(s - mn);
    l = l * corr + pr;
#pragma unroll
    for (int c = 0; c < 8; ++c) acc[c] = acc[c] * corr + pr * vv[j][c];
    m = mn;
  }
  float invl = 1.f / l;
#pragma unroll
  for (int c = 0; c < 8; ++c)
    yBuf[(((size_t)b * 192 + 128 + head * 8 + c) * 160 + t) * 160 + x] = acc[c] * invl;
}

// ---------------------------------------------------------------------------
// Host orchestration
// ---------------------------------------------------------------------------
extern "C" void kernel_launch(void* const* d_in, const int* in_sizes, int n_in,
                              void* d_out, int out_size, void* d_ws, size_t ws_size,
                              hipStream_t stream) {
  (void)in_sizes; (void)n_in; (void)out_size; (void)ws_size;
  const float* x      = (const float*)d_in[0];
  const float* w_in   = (const float*)d_in[1];
  const float* b_in   = (const float*)d_in[2];
  const float* w_out  = (const float*)d_in[3];
  const float* b_out  = (const float*)d_in[4];
  const float* lscale = (const float*)d_in[5];
  const float* lrscale= (const float*)d_in[6];
  const float* cpb_w1 = (const float*)d_in[7];
  const float* cpb_b1 = (const float*)d_in[8];
  const float* cpb_w2 = (const float*)d_in[9];
  float* out = (float*)d_out;

  char* ws = (char*)d_ws;
  size_t off = 0;
  auto carve = [&](size_t bytes) -> char* {
    char* p = ws + off;
    off = (off + bytes + 255) & ~(size_t)255;
    return p;
  };
  // xpad reused later as ypad (same size, 192 channels both times)
  unsigned short* xpad = (unsigned short*)carve((size_t)2*162*162*192 * 2);
  unsigned short* wA   = (unsigned short*)carve((size_t)384*1728 * 2);
  unsigned short* wB   = (unsigned short*)carve((size_t)192*1728 * 2);
  float* xp    = (float*)carve((size_t)2*384*160*160 * 4);
  float* biasT = (float*)carve((size_t)8*625 * 4);
  float* yBuf  = (float*)carve((size_t)2*192*160*160 * 4);
  float* vtmp  = (float*)carve((size_t)2*8*160*160*8 * 4);

  const long padTotal = (long)2 * 162 * 162 * 192;
  const int padBlocks = (int)((padTotal + 255) / 256);

  // 1) pack activations & weights to bf16
  pack_pad_nhwc_bf16<<<padBlocks, 256, 0, stream>>>(x, xpad, 192);
  pack_w_bf16<<<(384 * 1728 + 255) / 256, 256, 0, stream>>>(w_in, wA, 384);
  pack_w_bf16<<<(192 * 1728 + 255) / 256, 256, 0, stream>>>(w_out, wB, 192);

  // 2) conv_in: 192 -> 384 via WMMA implicit GEMM (6 M-blocks x 800 N-tiles)
  conv3x3_wmma_kernel<<<6 * 800, 128, 0, stream>>>(xpad, wA, b_in, xp, 384, 6);

  // 3) relative-position bias table
  cpb_bias_kernel<<<1, 128, 0, stream>>>(cpb_w1, cpb_b1, cpb_w2, biasT);

  // 4) window attentions (plain + shifted); 2048 windows, wave per head
  win_attn_kernel<<<2048, 256, 0, stream>>>(xp, biasT, lscale, yBuf, 0,   0,  0, 0);
  win_attn_kernel<<<2048, 256, 0, stream>>>(xp, biasT, lscale, yBuf, 128, 64, 3, 2);

  // 5) long-range attention: rows then columns
  lr_attn_rows<<<2 * 160 * 8, 160, 0, stream>>>(xp, lrscale, vtmp);
  lr_attn_cols<<<2 * 160 * 8, 160, 0, stream>>>(xp, lrscale, vtmp, yBuf);

  // 6) repack y (192 ch) into padded NHWC bf16 (reuse xpad) and conv_out
  pack_pad_nhwc_bf16<<<padBlocks, 256, 0, stream>>>(yBuf, xpad, 192);
  conv3x3_wmma_kernel<<<3 * 800, 128, 0, stream>>>(xpad, wB, b_out, out, 192, 3);
}
